// MultiHeadAttention_15341623181491
// MI455X (gfx1250) — matrix operand order repaired and checked by reading, not compiled, not run on hardware
//
#include <hip/hip_runtime.h>
#include <cstdint>
#include <cstddef>

// Problem constants (match reference)
#define B_ 4
#define S_ 2048
#define E_ 1024
#define H_ 16
#define D_ 64
#define M_ (B_ * S_)   // 8192 rows

typedef __attribute__((ext_vector_type(16))) _Float16 v16h;
typedef __attribute__((ext_vector_type(8)))  float    v8f;

union FragH {
    v16h     v;
    uint32_t u[8];
    _Float16 h[16];
};
union PairU {
    uint32_t u;
    _Float16 h[2];
};

// pack two f32 -> two f16 in one dword (v_cvt_pk_rtz_f16_f32)
__device__ __forceinline__ uint32_t pk_f16(float x, float y) {
    return __builtin_bit_cast(uint32_t, __builtin_amdgcn_cvt_pkrtz(x, y));
}

// A-fragment K index for 16-bit 16x32 A matrix (contiguous, matches B):
// VGPR j: K = 2j + 16*half
__device__ __forceinline__ int a_kidx(int j, int half) {
    return 2 * j + 16 * half;
}

// ---------------- DPP-based 16-lane reductions (VALU only, no LDS pipe) ----
template <int M>
__device__ __forceinline__ float dpp8_xor(float x) {
    constexpr int sel = ((0 ^ M)      ) | ((1 ^ M) <<  3) | ((2 ^ M) <<  6) |
                        ((3 ^ M) <<  9) | ((4 ^ M) << 12) | ((5 ^ M) << 15) |
                        ((6 ^ M) << 18) | ((7 ^ M) << 21);
    return __builtin_bit_cast(float,
        __builtin_amdgcn_mov_dpp8(__builtin_bit_cast(int, x), sel));
}
__device__ __forceinline__ float row_mirror(float x) {
    // DPP16 ROW_MIRROR (0x140): lane i <- lane 15-i within each row of 16
    return __builtin_bit_cast(float,
        __builtin_amdgcn_update_dpp(0, __builtin_bit_cast(int, x),
                                    0x140, 0xf, 0xf, true));
}
__device__ __forceinline__ float red16_max(float x) {
    x = fmaxf(x, dpp8_xor<1>(x));
    x = fmaxf(x, dpp8_xor<2>(x));
    x = fmaxf(x, dpp8_xor<4>(x));
    x = fmaxf(x, row_mirror(x));
    return x;
}
__device__ __forceinline__ float red16_sum(float x) {
    x += dpp8_xor<1>(x);
    x += dpp8_xor<2>(x);
    x += dpp8_xor<4>(x);
    x += row_mirror(x);
    return x;
}

// ---------------- optional gfx1250 async global->LDS copy path -------------
#if defined(__has_builtin)
#if __has_builtin(__builtin_amdgcn_global_load_async_to_lds_b32)
#define HAVE_ASYNC_LDS 1
#endif
#endif

typedef __attribute__((address_space(1))) int gint_t;   // global (device) AS
typedef __attribute__((address_space(3))) int lint_t;   // LDS AS

__device__ __forceinline__ void wait_asynccnt0() {
#if defined(__has_builtin) && __has_builtin(__builtin_amdgcn_s_wait_asynccnt)
    __builtin_amdgcn_s_wait_asynccnt(0);
#else
    asm volatile("s_wait_asynccnt 0" ::: "memory");
#endif
}

// ---------------------------------------------------------------------------
// GEMM: C[M,N] = A[M,K] @ W[N,K]^T + bias[N]
// A f32 or f16 (converted to f16 during LDS staging via v_cvt_pk_rtz_f16_f32).
// Block = 256 threads (8 waves). Block tile 128x128, wave tile 32x64,
// K staged in panels of 64 -> 16 v_wmma_f32_16x16x32_f16 per wave per panel.
// ---------------------------------------------------------------------------
template <typename TA, typename TO>
__global__ __launch_bounds__(256)
void gemm_bias_wmma(const TA* __restrict__ A, const float* __restrict__ W,
                    const float* __restrict__ bias, TO* __restrict__ C,
                    int Mdim, int Ndim, int Kdim)
{
    constexpr int SA = 72;                 // LDS row stride (f16), padded
    __shared__ _Float16 As[128 * SA];      // 128 rows x 64 K (f16)
    __shared__ _Float16 Bs[128 * SA];

    const int tid  = threadIdx.x;
    const int lane = tid & 31;
    const int wave = tid >> 5;             // 0..7
    const int half = lane >> 4;
    const int l16  = lane & 15;
    const int m0 = blockIdx.x * 128;
    const int n0 = blockIdx.y * 128;
    const int wm = (wave >> 1) * 32;       // wave M offset in tile (0..96)
    const int wn = (wave & 1) * 64;        // wave N offset in tile (0,64)

    v8f acc[2][4] = {};

    for (int k0 = 0; k0 < Kdim; k0 += 64) {
        __syncthreads();                   // previous stage fully consumed
        // Stage 128x64 A panel and 128x64 W panel as f16 (dword granules)
#pragma unroll
        for (int i = 0; i < 16; ++i) {
            int idx = tid + i * 256;       // 0..4095 dword granules
            int r = idx >> 5, c = (idx & 31) * 2;
            size_t ga = (size_t)(m0 + r) * Kdim + (k0 + c);
            size_t gw = (size_t)(n0 + r) * Kdim + (k0 + c);
            if constexpr (sizeof(TA) == 2) {
                *(uint32_t*)&As[r * SA + c] = *(const uint32_t*)&A[ga];
            } else {
                float2 av = *(const float2*)&A[ga];
                *(uint32_t*)&As[r * SA + c] = pk_f16(av.x, av.y);
            }
            float2 wv = *(const float2*)&W[gw];
            *(uint32_t*)&Bs[r * SA + c] = pk_f16(wv.x, wv.y);
            if (i == 0 && k0 + 64 < Kdim) {      // prefetch next K panel
                __builtin_prefetch(&A[ga + 64], 0, 1);
                __builtin_prefetch(&W[gw + 64], 0, 1);
            }
        }
        __syncthreads();

#pragma unroll
        for (int ks = 0; ks < 64; ks += 32) {
            FragH af[2], bf[4];
#pragma unroll
            for (int t = 0; t < 2; ++t) {
                const _Float16* ap = &As[(wm + t * 16 + l16) * SA + ks];
#pragma unroll
                for (int j = 0; j < 8; ++j)
                    af[t].u[j] = *(const uint32_t*)(ap + a_kidx(j, half));     // A: lane=row
            }
#pragma unroll
            for (int u = 0; u < 4; ++u) {
                const _Float16* bp = &Bs[(wn + u * 16 + l16) * SA + ks];
#pragma unroll
                for (int j = 0; j < 8; ++j)
                    bf[u].u[j] = *(const uint32_t*)(bp + (2 * j + 16 * half)); // B: lane=col (W^T)
            }
#pragma unroll
            for (int t = 0; t < 2; ++t)
#pragma unroll
                for (int u = 0; u < 4; ++u)
                    acc[t][u] = __builtin_amdgcn_wmma_f32_16x16x32_f16(
                        false, af[t].v, false, bf[u].v, (short)0, acc[t][u], false, false);
        }
    }

    // Epilogue: C-frag lane=column n, rows m = j + 8*half (coalesced across lanes)
#pragma unroll
    for (int t = 0; t < 2; ++t)
#pragma unroll
        for (int u = 0; u < 4; ++u) {
            int n = n0 + wn + u * 16 + l16;
            float bb = bias[n];
#pragma unroll
            for (int j = 0; j < 8; ++j) {
                int m = m0 + wm + t * 16 + j + 8 * half;
                C[(size_t)m * Ndim + n] = (TO)(acc[t][u][j] + bb);
            }
        }
}

// ---------------------------------------------------------------------------
// Flash attention: block = 4 waves sharing one (b,h); each wave owns 16
// queries x D=64. K/V tiles (32 keys) staged cooperatively into LDS once per
// block; K via async global->LDS copies (ASYNCcnt) when available, V staged
// transposed so PV B-fragments are contiguous dword loads.
// Softmax in exp2 domain; row reductions via DPP (no LDS traffic).
// ---------------------------------------------------------------------------
__global__ __launch_bounds__(128)
void attn_flash_wmma(const _Float16* __restrict__ Q, const _Float16* __restrict__ K,
                     const _Float16* __restrict__ V, _Float16* __restrict__ Ctx)
{
    constexpr int SK = 72;                        // Ks row stride (f16)
    constexpr int SV = 40;                        // Vt row stride (f16)
    constexpr int PS = 40;                        // P row stride (f16)
    __shared__ _Float16 Ks[32 * SK];              // [k in tile][d]
    __shared__ _Float16 Vt[64 * SV];              // [d][k in tile] (transposed)
    __shared__ _Float16 Plds[4][16 * PS];         // per-wave private P tile

    const int tid  = threadIdx.x;
    const int lane = tid & 31;
    const int wave = tid >> 5;
    const int half = lane >> 4;
    const int l16  = lane & 15;

    const int bh = blockIdx.y;                    // 0..B*H-1
    const int bb = bh >> 4, hh = bh & 15;
    const int q0 = blockIdx.x * 64 + wave * 16;   // this wave's 16 queries

    const _Float16* Qh = Q + (size_t)bb * S_ * E_ + hh * D_;
    const _Float16* Kh = K + (size_t)bb * S_ * E_ + hh * D_;
    const _Float16* Vh = V + (size_t)bb * S_ * E_ + hh * D_;

    // Q A-fragments for the two 32-wide halves of D=64 (lane = query row)
    FragH aq[2];
#pragma unroll
    for (int dh = 0; dh < 2; ++dh) {
        const _Float16* qp = Qh + (size_t)(q0 + l16) * E_ + dh * 32;
#pragma unroll
        for (int j = 0; j < 8; ++j)
            aq[dh].u[j] = *(const uint32_t*)(qp + a_kidx(j, half));
    }

    // softmax in exp2 domain: fold 1/sqrt(64) * log2(e) into one scale
    const float SCL = 0.125f * 1.44269504088896340736f;

    v8f acc[4] = {};                      // O accumulator: 4 d-tiles of 16
    float mrow[8], lrow[8];               // running max / sum (rows j + 8*half)
#pragma unroll
    for (int j = 0; j < 8; ++j) { mrow[j] = -3.0e38f; lrow[j] = 0.0f; }

    for (int kt = 0; kt < S_; kt += 32) {
        __syncthreads();                  // previous tile fully consumed
        // ---- cooperative staging: K natural (async if available), V transposed
#pragma unroll
        for (int i = 0; i < 8; ++i) {
            int idx = tid + i * 128;      // 0..1023 dword granules (32x64 tile)
            int r = idx >> 5, c = (idx & 31) * 2;
#if HAVE_ASYNC_LDS
            __builtin_amdgcn_global_load_async_to_lds_b32(
                (gint_t*)&Kh[(size_t)(kt + r) * E_ + c],
                (lint_t*)&Ks[r * SK + c],
                0, 0);
#else
            *(uint32_t*)&Ks[r * SK + c] =
                *(const uint32_t*)&Kh[(size_t)(kt + r) * E_ + c];
#endif
            PairU pv;
            pv.u = *(const uint32_t*)&Vh[(size_t)(kt + r) * E_ + c];
            Vt[(c + 0) * SV + r] = pv.h[0];
            Vt[(c + 1) * SV + r] = pv.h[1];
        }
#if HAVE_ASYNC_LDS
        wait_asynccnt0();                 // this wave's async copies landed
#endif
        __syncthreads();

        // ---- S = (Q K^T): two 16x16 f32 frags over this 32-key tile ----
        v8f sfr[2] = {};
#pragma unroll
        for (int c = 0; c < 2; ++c) {
            const _Float16* kp = &Ks[(c * 16 + l16) * SK];   // lane = key column
#pragma unroll
            for (int dh = 0; dh < 2; ++dh) {
                FragH bk;
#pragma unroll
                for (int j = 0; j < 8; ++j)
                    bk.u[j] = *(const uint32_t*)(kp + (2 * j + 16 * half + dh * 32));
                sfr[c] = __builtin_amdgcn_wmma_f32_16x16x32_f16(
                    false, aq[dh].v, false, bk.v, (short)0, sfr[c], false, false);
            }
        }

        // ---- online softmax (rows across the 16 lanes of each half-wave) ----
        float tm[8], alpha[8], rs[8];
#pragma unroll
        for (int j = 0; j < 8; ++j) {
            sfr[0][j] *= SCL;
            sfr[1][j] *= SCL;
            tm[j] = red16_max(fmaxf(sfr[0][j], sfr[1][j]));
        }
#pragma unroll
        for (int j = 0; j < 8; ++j) {
            float mn = fmaxf(mrow[j], tm[j]);
            alpha[j] = __builtin_amdgcn_exp2f(mrow[j] - mn);
            mrow[j]  = mn;
            sfr[0][j] = __builtin_amdgcn_exp2f(sfr[0][j] - mn);
            sfr[1][j] = __builtin_amdgcn_exp2f(sfr[1][j] - mn);
            rs[j] = red16_sum(sfr[0][j] + sfr[1][j]);
        }
#pragma unroll
        for (int j = 0; j < 8; ++j) lrow[j] = lrow[j] * alpha[j] + rs[j];
#pragma unroll
        for (int t = 0; t < 4; ++t)
#pragma unroll
            for (int j = 0; j < 8; ++j) acc[t][j] *= alpha[j];

        // ---- P: C-frag layout -> LDS -> A-frag layout (wave-private tile) ----
        _Float16* pl = Plds[wave];
#pragma unroll
        for (int c = 0; c < 2; ++c)
#pragma unroll
            for (int j = 0; j < 8; ++j)
                pl[(j + 8 * half) * PS + (c * 16 + l16)] = (_Float16)sfr[c][j];
        asm volatile("s_wait_dscnt 0" ::: "memory");  // LDS in-order per wave

        FragH pa;
        {
            const _Float16* pp = pl + l16 * PS;
#pragma unroll
            for (int j = 0; j < 8; ++j)
                pa.u[j] = *(const uint32_t*)(pp + a_kidx(j, half));
        }

        // ---- O += P · V : 4 WMMAs, B-frags from transposed V in LDS ----
#pragma unroll
        for (int dt = 0; dt < 4; ++dt) {
            FragH bv;
            const _Float16* vp = &Vt[(dt * 16 + l16) * SV];  // lane = d column
#pragma unroll
            for (int j = 0; j < 8; ++j)
                bv.u[j] = *(const uint32_t*)(vp + (2 * j + 16 * half));
            acc[dt] = __builtin_amdgcn_wmma_f32_16x16x32_f16(
                false, pa.v, false, bv.v, (short)0, acc[dt], false, false);
        }
    }

    // ---- finalize: O *= 1/l ; write f16 context [b*S+q, h*64+d] ----
    float inv[8];
#pragma unroll
    for (int j = 0; j < 8; ++j) inv[j] = __builtin_amdgcn_rcpf(lrow[j]);
#pragma unroll
    for (int dt = 0; dt < 4; ++dt) {
        const int d = hh * D_ + dt * 16 + l16;
#pragma unroll
        for (int j = 0; j < 8; ++j) {
            const int q = q0 + j + 8 * half;
            Ctx[((size_t)bb * S_ + q) * E_ + d] = (_Float16)(acc[dt][j] * inv[j]);
        }
    }
}

// ---------------------------------------------------------------------------
extern "C" void kernel_launch(void* const* d_in, const int* in_sizes, int n_in,
                              void* d_out, int out_size, void* d_ws, size_t ws_size,
                              hipStream_t stream)
{
    (void)in_sizes; (void)n_in; (void)out_size; (void)ws_size;

    const float* x  = (const float*)d_in[0];
    const float* Wq = (const float*)d_in[1];
    const float* bq = (const float*)d_in[2];
    const float* Wk = (const float*)d_in[3];
    const float* bk = (const float*)d_in[4];
    const float* Wv = (const float*)d_in[5];
    const float* bv = (const float*)d_in[6];
    const float* Wo = (const float*)d_in[7];
    const float* bo = (const float*)d_in[8];
    float* out = (float*)d_out;

    // Workspace: Q, K, V, ctx as f16 [M_, E_]  (4 x 16 MiB = 64 MiB)
    _Float16* Qb = (_Float16*)d_ws;
    _Float16* Kb = Qb + (size_t)M_ * E_;
    _Float16* Vb = Kb + (size_t)M_ * E_;
    _Float16* Cb = Vb + (size_t)M_ * E_;

    dim3 gemm_blk(256);
    dim3 gg(M_ / 128, E_ / 128);          // 64 x 8 blocks per GEMM

    gemm_bias_wmma<float, _Float16><<<gg, gemm_blk, 0, stream>>>(x, Wq, bq, Qb, M_, E_, E_);
    gemm_bias_wmma<float, _Float16><<<gg, gemm_blk, 0, stream>>>(x, Wk, bk, Kb, M_, E_, E_);
    gemm_bias_wmma<float, _Float16><<<gg, gemm_blk, 0, stream>>>(x, Wv, bv, Vb, M_, E_, E_);

    dim3 ga(S_ / 64, B_ * H_);            // 32 x 64 blocks
    dim3 attn_blk(128);
    attn_flash_wmma<<<ga, attn_blk, 0, stream>>>(Qb, Kb, Vb, Cb);

    gemm_bias_wmma<_Float16, float><<<gg, gemm_blk, 0, stream>>>(Cb, Wo, bo, out, M_, E_, E_);
}
